// MultiheadAttention_71494025610098
// MI455X (gfx1250) — matrix operand order repaired and checked by reading, not compiled, not run on hardware
//
#include <hip/hip_runtime.h>
#include <hip/hip_bf16.h>

// Problem constants (match reference)
#define E_DIM 1024
#define H_NUM 16
#define D_HEAD 64
#define B_SZ 2
#define L_SEQ 2048
#define BH (B_SZ * H_NUM)   // 32
#define BL (B_SZ * L_SEQ)   // 4096
#define E3 (3 * E_DIM)      // 3072

typedef __bf16 bf16x16 __attribute__((ext_vector_type(16)));
typedef float  floatx8 __attribute__((ext_vector_type(8)));

union Frag16 { bf16x16 v; unsigned int u[8]; };

__device__ __forceinline__ unsigned short f2bf(float f) {
  unsigned int u = __float_as_uint(f);
  u += 0x7fffu + ((u >> 16) & 1u);   // round-to-nearest-even
  return (unsigned short)(u >> 16);
}

__device__ __forceinline__ floatx8 wmma_bf16(bf16x16 a, bf16x16 b, floatx8 c) {
  // 8 args: (neg_a, A, neg_b, B, c_mod, C, reuse_a, reuse_b)
  return __builtin_amdgcn_wmma_f32_16x16x32_bf16(false, a, false, b, (short)0, c,
                                                 false, false);
}

// A fragment: 16(M) x 32(K) bf16 tile from row-major [*, ld].
// ISA layout: lane&15 = M; lane>=16 selects K+8; VGPR v: K = (v>>2)*16 + (v&3)*2.
__device__ __forceinline__ bf16x16 load_a_frag(const unsigned short* base,
                                               int ld, int m0, int k0, int lane) {
  Frag16 f;
  const unsigned short* p =
      base + (size_t)(m0 + (lane & 15)) * ld + k0 + (((lane >> 4) & 1) << 3);
#pragma unroll
  for (int v = 0; v < 8; ++v) {
    int k = ((v >> 2) << 4) + ((v & 3) << 1);
    f.u[v] = *(const unsigned int*)(p + k);
  }
  return f.v;
}

// B fragment: 32(K) x 16(N) bf16 tile from row-major [K, ld] source.
// lane&15 = N; lane>=16 selects K+8; VGPR v: K = (v>>2)*16 + (v&3)*2 (pair K, K+1).
__device__ __forceinline__ bf16x16 load_b_frag(const unsigned short* __restrict__ base,
                                               int ld, int k0, int n0, int lane) {
  Frag16 f;
  const unsigned short* p = base + (size_t)(k0 + (((lane >> 4) & 1) << 3)) * ld + n0 + (lane & 15);
#pragma unroll
  for (int v = 0; v < 8; ++v) {
    int k = ((v >> 2) << 4) + ((v & 3) << 1);
    f.u[v] = (unsigned int)p[(size_t)k * ld] | ((unsigned int)p[(size_t)(k + 1) * ld] << 16);
  }
  return f.v;
}

// ---------------------------------------------------------------------------
// fp32 -> bf16 bulk convert
__global__ void cvt_bf16_kernel(const float* __restrict__ src,
                                unsigned short* __restrict__ dst, int n) {
  int i = blockIdx.x * blockDim.x + threadIdx.x;
  if (i < n) dst[i] = f2bf(src[i]);
}

// ---------------------------------------------------------------------------
// QKV projection: [4096,1024] x [1024,3072] + bias.
// Writes Q row-major per (b,h), K transposed (KT[bh][d][l]), V row-major.
__global__ void qkv_gemm_kernel(const unsigned short* __restrict__ xb,
                                const unsigned short* __restrict__ wb,
                                const float* __restrict__ bias,
                                unsigned short* __restrict__ qb,
                                unsigned short* __restrict__ ktb,
                                unsigned short* __restrict__ vb) {
  const int lane = threadIdx.x & 31;
  const int wave = threadIdx.x >> 5;
  const int tile = blockIdx.x * 8 + wave;      // 12288 tiles total
  const int TN = E3 / 64;                      // 48
  const int tm = tile / TN, tn = tile % TN;
  const int m0 = tm * 16, n0 = tn * 64;

  floatx8 acc[4] = {{}, {}, {}, {}};
  for (int k0 = 0; k0 < E_DIM; k0 += 32) {
    bf16x16 a = load_a_frag(xb, E_DIM, m0, k0, lane);
#pragma unroll
    for (int nb = 0; nb < 4; ++nb)
      acc[nb] = wmma_bf16(a, load_b_frag(wb, E3, k0, n0 + nb * 16, lane), acc[nb]);
  }

  const int rowbase = m0 + ((lane >> 4) << 3);
  const int col = lane & 15;
#pragma unroll
  for (int nb = 0; nb < 4; ++nb) {
    const int n = n0 + nb * 16 + col;
    // reference column mapping: n -> (h, which in {q,k,v}, d)
    const int h = n / 192, rem = n % 192, which = rem >> 6, d = rem & 63;
    const float bv = bias[n];
#pragma unroll
    for (int r = 0; r < 8; ++r) {
      const int m = rowbase + r;
      const int b = m >> 11, l = m & (L_SEQ - 1);
      const int bh = b * H_NUM + h;
      const unsigned short val = f2bf(acc[nb][r] + bv);
      if (which == 0)
        qb[((size_t)bh * L_SEQ + l) * D_HEAD + d] = val;
      else if (which == 1)
        ktb[((size_t)bh * D_HEAD + d) * L_SEQ + l] = val;   // transposed store
      else
        vb[((size_t)bh * L_SEQ + l) * D_HEAD + d] = val;
    }
  }
}

// ---------------------------------------------------------------------------
// Fused QK^T + softmax + P@V.
// Pass 1: online row max/sum over 128 j-tiles (S recomputed, never spilled).
// Pass 2: recompute S, normalize -> single coalesced attention write, AND
//         transpose the P tile through a per-wave LDS buffer (C-layout ->
//         A-layout) to immediately accumulate O = P@V with WMMA. This removes
//         the 537 MB attention re-read a separate PV kernel would need.
__global__ void attn_fused_kernel(const unsigned short* __restrict__ qb,
                                  const unsigned short* __restrict__ ktb,
                                  const unsigned short* __restrict__ vb,
                                  float* __restrict__ attn,
                                  unsigned short* __restrict__ valb) {
  __shared__ unsigned short psmem[8 * 16 * 32];   // 1 KB per wave
  const int lane = threadIdx.x & 31;
  const int wave = threadIdx.x >> 5;
  const int tile = blockIdx.x * 8 + wave;      // bh*128 + ti, 4096 total
  const int ti = tile & 127;
  const int bh = tile >> 7;
  const int i0 = ti * 16;

  const unsigned short* q  = qb  + (size_t)bh * L_SEQ * D_HEAD;
  const unsigned short* kt = ktb + (size_t)bh * D_HEAD * L_SEQ;
  const unsigned short* v  = vb  + (size_t)bh * L_SEQ * D_HEAD;
  float* arow = attn + (size_t)bh * L_SEQ * L_SEQ;
  unsigned short* pt = &psmem[wave * 16 * 32];

  // Q fragments are invariant over j: hoist (2 chunks of K=32 cover D=64).
  const bf16x16 a0 = load_a_frag(q, D_HEAD, i0, 0, lane);
  const bf16x16 a1 = load_a_frag(q, D_HEAD, i0, 32, lane);

  float rm[8], rs[8];
#pragma unroll
  for (int r = 0; r < 8; ++r) { rm[r] = -3.4028235e38f; rs[r] = 0.0f; }

  // Pass 1: streaming max/sum per row (row = r + 8*(lane>=16); cols across 16 lanes)
  for (int jt = 0; jt < 128; ++jt) {
    const int j0 = jt * 16;
    floatx8 acc = {};
    acc = wmma_bf16(a0, load_b_frag(kt, L_SEQ, 0, j0, lane), acc);
    acc = wmma_bf16(a1, load_b_frag(kt, L_SEQ, 32, j0, lane), acc);
#pragma unroll
    for (int r = 0; r < 8; ++r) {
      float s = acc[r] * 0.125f;                       // 1/sqrt(64)
      float tm = s;
      tm = fmaxf(tm, __shfl_xor(tm, 1, 32));
      tm = fmaxf(tm, __shfl_xor(tm, 2, 32));
      tm = fmaxf(tm, __shfl_xor(tm, 4, 32));
      tm = fmaxf(tm, __shfl_xor(tm, 8, 32));           // row max within tile
      const float mnew = fmaxf(rm[r], tm);
      float e = __expf(s - mnew);
      e += __shfl_xor(e, 1, 32);
      e += __shfl_xor(e, 2, 32);
      e += __shfl_xor(e, 4, 32);
      e += __shfl_xor(e, 8, 32);                       // row sum within tile
      rs[r] = rs[r] * __expf(rm[r] - mnew) + e;
      rm[r] = mnew;
    }
  }
  float inv[8];
#pragma unroll
  for (int r = 0; r < 8; ++r) inv[r] = 1.0f / rs[r];

  // Pass 2: process pairs of 16-wide j-tiles (one K=32 chunk for P@V).
  floatx8 acco[4] = {{}, {}, {}, {}};
  const int rowoff = (lane >> 4) << 3;
  const int col = lane & 15;
  for (int jp = 0; jp < 64; ++jp) {
    const int j0 = jp * 32;
#pragma unroll
    for (int t = 0; t < 2; ++t) {
      const int jt0 = j0 + t * 16;
      floatx8 acc = {};
      acc = wmma_bf16(a0, load_b_frag(kt, L_SEQ, 0, jt0, lane), acc);
      acc = wmma_bf16(a1, load_b_frag(kt, L_SEQ, 32, jt0, lane), acc);
#pragma unroll
      for (int r = 0; r < 8; ++r) {
        const float p = __expf(acc[r] * 0.125f - rm[r]) * inv[r];
        arow[(size_t)(i0 + rowoff + r) * L_SEQ + jt0 + col] = p;   // attention out
        pt[(rowoff + r) * 32 + t * 16 + col] = f2bf(p);            // LDS stage
      }
    }
    __builtin_amdgcn_wave_barrier();   // keep store->load order for the backend
    // Reload staged P as an A fragment (cross-lane transpose via LDS).
    const bf16x16 ap = load_a_frag(pt, 32, 0, 0, lane);
#pragma unroll
    for (int nb = 0; nb < 4; ++nb)
      acco[nb] = wmma_bf16(ap, load_b_frag(v, D_HEAD, j0, nb * 16, lane), acco[nb]);
  }

  // Store O block: values[b, l, h*64+d]  (transpose folded into indexing)
  const int b = bh >> 4, h = bh & 15;
#pragma unroll
  for (int nb = 0; nb < 4; ++nb)
#pragma unroll
    for (int r = 0; r < 8; ++r) {
      const int l = i0 + rowoff + r;
      valb[((size_t)b * L_SEQ + l) * E_DIM + h * D_HEAD + nb * 16 + col] =
          f2bf(acco[nb][r]);
    }
}

// ---------------------------------------------------------------------------
// Output projection: o = values @ W_o + b_o  ([4096,1024] x [1024,1024], fp32 out)
__global__ void out_gemm_kernel(const unsigned short* __restrict__ valb,
                                const unsigned short* __restrict__ wob,
                                const float* __restrict__ bias,
                                float* __restrict__ out) {
  const int lane = threadIdx.x & 31;
  const int wave = threadIdx.x >> 5;
  const int tile = blockIdx.x * 8 + wave;      // 256 x 16 = 4096 tiles
  const int tm = tile >> 4, tn = tile & 15;
  const int m0 = tm * 16, n0 = tn * 64;

  floatx8 acc[4] = {{}, {}, {}, {}};
  for (int k0 = 0; k0 < E_DIM; k0 += 32) {
    bf16x16 a = load_a_frag(valb, E_DIM, m0, k0, lane);
#pragma unroll
    for (int nb = 0; nb < 4; ++nb)
      acc[nb] = wmma_bf16(a, load_b_frag(wob, E_DIM, k0, n0 + nb * 16, lane), acc[nb]);
  }

  const int rowoff = (lane >> 4) << 3;
  const int col = lane & 15;
#pragma unroll
  for (int nb = 0; nb < 4; ++nb) {
    const int n = n0 + nb * 16 + col;
    const float bv = bias[n];
#pragma unroll
    for (int r = 0; r < 8; ++r)
      out[(size_t)(m0 + rowoff + r) * E_DIM + n] = acc[nb][r] + bv;
  }
}

// ---------------------------------------------------------------------------
extern "C" void kernel_launch(void* const* d_in, const int* in_sizes, int n_in,
                              void* d_out, int out_size, void* d_ws, size_t ws_size,
                              hipStream_t stream) {
  (void)in_sizes; (void)n_in; (void)out_size; (void)ws_size;
  const float* x    = (const float*)d_in[0];
  const float* Wqkv = (const float*)d_in[1];
  const float* bqkv = (const float*)d_in[2];
  const float* Wo   = (const float*)d_in[3];
  const float* bo   = (const float*)d_in[4];

  float* out  = (float*)d_out;                       // [B, L, E] fp32
  float* attn = out + (size_t)BL * E_DIM;            // [B, H, L, L] fp32

  // Workspace layout (bf16 = unsigned short), total ~48 MB
  unsigned short* ws    = (unsigned short*)d_ws;
  unsigned short* xb    = ws;                                    // BL*E
  unsigned short* wqkvb = xb    + (size_t)BL * E_DIM;            // E*3E
  unsigned short* wob   = wqkvb + (size_t)E_DIM * E3;            // E*E
  unsigned short* qb    = wob   + (size_t)E_DIM * E_DIM;         // BH*L*D
  unsigned short* ktb   = qb    + (size_t)BH * L_SEQ * D_HEAD;   // BH*D*L
  unsigned short* vb    = ktb   + (size_t)BH * L_SEQ * D_HEAD;   // BH*L*D
  unsigned short* valb  = vb    + (size_t)BH * L_SEQ * D_HEAD;   // BL*E

  cvt_bf16_kernel<<<(BL * E_DIM) / 256, 256, 0, stream>>>(x, xb, BL * E_DIM);
  cvt_bf16_kernel<<<(E_DIM * E3) / 256, 256, 0, stream>>>(Wqkv, wqkvb, E_DIM * E3);
  cvt_bf16_kernel<<<(E_DIM * E_DIM) / 256, 256, 0, stream>>>(Wo, wob, E_DIM * E_DIM);

  qkv_gemm_kernel<<<1536, 256, 0, stream>>>(xb, wqkvb, bqkv, qb, ktb, vb);
  attn_fused_kernel<<<512, 256, 0, stream>>>(qb, ktb, vb, attn, valb);
  out_gemm_kernel<<<512, 256, 0, stream>>>(valb, wob, bo, out);
}
